// EFL_59863254172421
// MI455X (gfx1250) — matrix operand order repaired and checked by reading, not compiled, not run on hardware
//
#include <hip/hip_runtime.h>
#include <math.h>

typedef __bf16 bf16_t;
typedef __attribute__((ext_vector_type(16))) __bf16 v16bf;
typedef __attribute__((ext_vector_type(8)))  __bf16 v8bf;
typedef __attribute__((ext_vector_type(8)))  float  v8f;

#define EE    2304
#define HH    12
#define DD    192
#define LW    256
#define FFD   3072
#define MPE_  7200
#define BB    2
#define FRN   2047
#define SPL   2048
#define NTOK  (BB*SPL)      // 4096
#define MB    16            // q-rows per band-attention block
#define SCALEQ 0.07216878364870322f   // 1/sqrt(192)

#define SHUF16(lo,hi) __builtin_shufflevector(lo,hi,0,1,2,3,4,5,6,7,8,9,10,11,12,13,14,15)

__device__ __forceinline__ float gelu_f(float x) {
    return 0.5f * x * (1.0f + erff(x * 0.70710678118654752f));
}

__device__ __forceinline__ float dot192f(const float* __restrict__ a,
                                         const float* __restrict__ b) {
    float s = 0.f;
#pragma unroll 8
    for (int i = 0; i < DD; ++i) s = fmaf(a[i], b[i], s);
    return s;
}

// --- CDNA5 async global->LDS copy (ISA 15.18.3, tracked with ASYNCcnt) -----
__device__ __forceinline__ void async_ld_b128(unsigned lds_off, const void* gaddr) {
    asm volatile("global_load_async_to_lds_b128 %0, %1, off"
                 :: "v"(lds_off), "v"(gaddr) : "memory");
}
__device__ __forceinline__ void wait_async0() {
    asm volatile("s_wait_asynccnt 0" ::: "memory");
}

// ---------------------------------------------------------------------------
// fp32 -> bf16 elementwise
// ---------------------------------------------------------------------------
__global__ void k_f32_to_bf16(const float* __restrict__ x, bf16_t* __restrict__ y, long n) {
    long i = (long)blockIdx.x * blockDim.x + threadIdx.x;
    long stride = (long)gridDim.x * blockDim.x;
    for (; i < n; i += stride) y[i] = (bf16_t)x[i];
}

// ---------------------------------------------------------------------------
// Weight fp32 (KxN row-major) -> bf16 in WMMA B-fragment-major layout.
// ---------------------------------------------------------------------------
__global__ void k_w_to_frag(const float* __restrict__ w, bf16_t* __restrict__ dst,
                            int K, int N) {
    long n = (long)K * N;
    long stride = (long)gridDim.x * blockDim.x;
    int gridN = N >> 7;
    for (long i = (long)blockIdx.x * blockDim.x + threadIdx.x; i < n; i += stride) {
        int kk  = (int)(i / N);
        int col = (int)(i % N);
        int ks = kk >> 5, kr = kk & 31;
        int nb = col >> 7, nc = col & 127;
        int nt = nc >> 4;
        int lane = (nc & 15) + 16 * (kr >> 4);
        int e = kr & 15;
        long d = (((long)ks * gridN + nb) << 12) + nt * 512 + lane * 16 + e;
        dst[d] = (bf16_t)w[i];
    }
}

// ---------------------------------------------------------------------------
// Embedding
// ---------------------------------------------------------------------------
__global__ void k_embed(const float* __restrict__ feat, const float* __restrict__ cls,
                        const float* __restrict__ pos_emb, const float* __restrict__ tok_emb,
                        const int* __restrict__ pids, float* __restrict__ h) {
    long n = (long)BB * SPL * EE;
    long stride = (long)gridDim.x * blockDim.x;
    for (long i = (long)blockIdx.x * blockDim.x + threadIdx.x; i < n; i += stride) {
        int e = (int)(i % EE);
        long r = i / EE;
        int t = (int)(r % SPL);
        int b = (int)(r / SPL);
        float xv; int pid, tok;
        if (t == 0)        { xv = cls[e];  pid = MPE_ - 2; tok = 1; }
        else if (t <= FRN) { xv = feat[((long)b * FRN + (t - 1)) * EE + e];
                             pid = pids[b * FRN + t - 1] % (MPE_ - 2); tok = 0; }
        else               { xv = 0.f; pid = MPE_ - 1; tok = 0; }
        h[i] = xv + pos_emb[(long)pid * EE + e] + tok_emb[(long)tok * EE + e];
    }
}

// ---------------------------------------------------------------------------
// LayerNorm (+optional residual)
// ---------------------------------------------------------------------------
__global__ __launch_bounds__(256)
void k_ln(const float* __restrict__ x, long xr,
          const float* __restrict__ res, long rr,
          const float* __restrict__ g, const float* __restrict__ bb,
          float* __restrict__ y, long yr,
          bf16_t* __restrict__ ybf, long br, int n) {
    long row = blockIdx.x;
    const float* xp = x + row * xr;
    const float* rp = res ? res + row * rr : nullptr;
    float s = 0.f, q = 0.f;
    for (int c = threadIdx.x; c < n; c += 256) {
        float v = xp[c] + (rp ? rp[c] : 0.f);
        s += v; q += v * v;
    }
    __shared__ float sh[512];
    sh[threadIdx.x] = s; sh[256 + threadIdx.x] = q;
    __syncthreads();
    for (int off = 128; off > 0; off >>= 1) {
        if (threadIdx.x < (unsigned)off) {
            sh[threadIdx.x]       += sh[threadIdx.x + off];
            sh[256 + threadIdx.x] += sh[256 + threadIdx.x + off];
        }
        __syncthreads();
    }
    float mean = sh[0] / n;
    float var  = sh[256] / n - mean * mean;
    float rstd = rsqrtf(var + 1e-5f);
    for (int c = threadIdx.x; c < n; c += 256) {
        float v = xp[c] + (rp ? rp[c] : 0.f);
        float o = (v - mean) * rstd * g[c] + bb[c];
        if (y)   y[row * yr + c]   = o;
        if (ybf) ybf[row * br + c] = (bf16_t)o;
    }
}

// ---------------------------------------------------------------------------
// WMMA bf16 GEMM. A staged via async global->LDS (double buffered, ASYNCcnt);
// B fragment-major in global, loaded directly as 32B fragments.
// ---------------------------------------------------------------------------
__global__ __launch_bounds__(256)
void k_wmma_gemm(const bf16_t* __restrict__ A, long lda,
                 const bf16_t* __restrict__ Bf, int gridN,
                 const float* __restrict__ bias,
                 float* __restrict__ C, bf16_t* __restrict__ Cbf, long ldc,
                 int K, float alpha, int act) {
    __shared__ __align__(16) bf16_t As[2][128 * 32];
    const int tid  = threadIdx.x;
    const int wave = tid >> 5;
    const int lane = tid & 31;
    const int m0 = blockIdx.y * 128;
    const int wm = (wave & 3) * 32;
    const int wn = (wave >> 2) * 64;
    const int half = lane >> 4;
    const int lc   = lane & 15;
    const int nsteps = K >> 5;

    const unsigned ldsbase = (unsigned)(uintptr_t)&As[0][0];   // low 32 bits = LDS offset
    const int j0 = tid,       r0 = j0 >> 2, c0 = j0 & 3;
    const int j1 = tid + 256, r1 = j1 >> 2, c1 = j1 & 3;

    // stage k-step 0 asynchronously
    async_ld_b128(ldsbase + j0 * 16, A + (long)(m0 + r0) * lda + c0 * 8);
    async_ld_b128(ldsbase + j1 * 16, A + (long)(m0 + r1) * lda + c1 * 8);
    wait_async0();
    __syncthreads();

    v8f acc[2][4] = {};
    int cur = 0;
    for (int ks = 0; ks < nsteps; ++ks) {
        if (ks + 1 < nsteps) {   // async-stage next A tile into the other buffer
            unsigned lb = ldsbase + (cur ^ 1) * 8192;
            async_ld_b128(lb + j0 * 16, A + (long)(m0 + r0) * lda + (ks + 1) * 32 + c0 * 8);
            async_ld_b128(lb + j1 * 16, A + (long)(m0 + r1) * lda + (ks + 1) * 32 + c1 * 8);
            __builtin_prefetch(Bf + (((long)(ks + 1) * gridN + blockIdx.x) << 12) + tid * 32, 0, 0);
        }

        const bf16_t* bbase = Bf + ((((long)ks * gridN + blockIdx.x) << 12)
                                    + (long)(lc + 16 * half) * 16);
        v16bf af[2], bfr[4];
#pragma unroll
        for (int mt = 0; mt < 2; ++mt) {
            int row = wm + mt * 16 + lc;
            v8bf lo = *(const v8bf*)&As[cur][row * 32 + 16 * half];
            v8bf hi = *(const v8bf*)&As[cur][row * 32 + 16 * half + 8];
            af[mt] = SHUF16(lo, hi);
        }
#pragma unroll
        for (int nt = 0; nt < 4; ++nt)
            bfr[nt] = *(const v16bf*)(bbase + ((wn >> 4) + nt) * 512);
#pragma unroll
        for (int mt = 0; mt < 2; ++mt)
#pragma unroll
            for (int nt = 0; nt < 4; ++nt)
                acc[mt][nt] = __builtin_amdgcn_wmma_f32_16x16x32_bf16(
                    false, af[mt], false, bfr[nt], (short)0, acc[mt][nt], false, false);

        if (ks + 1 < nsteps) wait_async0();
        __syncthreads();
        cur ^= 1;
    }

#pragma unroll
    for (int mt = 0; mt < 2; ++mt) {
#pragma unroll
        for (int nt = 0; nt < 4; ++nt) {
#pragma unroll
            for (int r = 0; r < 8; ++r) {
                int row = m0 + wm + mt * 16 + half * 8 + r;
                int col = blockIdx.x * 128 + wn + nt * 16 + lc;
                float v = acc[mt][nt][r];
                if (bias) v += bias[col];
                v *= alpha;
                if (act) v = gelu_f(v);
                if (C)   C[(long)row * ldc + col] = v;
                if (Cbf) Cbf[(long)row * ldc + col] = (bf16_t)v;
            }
        }
    }
}

// ---------------------------------------------------------------------------
// Small GEMM (head / qg)
// ---------------------------------------------------------------------------
__global__ void k_small_gemm(const float* __restrict__ x, long xs,
                             const float* __restrict__ w, const float* __restrict__ bias,
                             float* __restrict__ y, long ys,
                             int M, int N, int K, float alpha, int act) {
    long i = (long)blockIdx.x * blockDim.x + threadIdx.x;
    if (i >= (long)M * N) return;
    int m = (int)(i / N), n = (int)(i % N);
    float s = 0.f;
    for (int e = 0; e < K; ++e) s = fmaf(x[m * xs + e], w[(long)e * N + n], s);
    s = alpha * (s + (bias ? bias[n] : 0.f));
    if (act) s = gelu_f(s);
    y[m * ys + n] = s;
}

// ---------------------------------------------------------------------------
// Banded local attention, full WMMA (scores AND P*V).
// Block = (b,h,chunk,16 q-rows), 384 threads (12 waves), dynamic LDS:
//   sc[16][769] f32 (49216B) | Qs[16*192] bf16 (6144B) | Vt[192*32] bf16 (12288B)
// Scores: each wave computes 16x64 of the 16x768 band tile.
// P*V: per 32-key step, V tile staged transposed in LDS; each wave owns one
// 16-wide D column tile; A-frag = probabilities converted f32->bf16 from sc.
// ---------------------------------------------------------------------------
#define SC_BYTES (MB*(3*LW+1)*4)
#define QS_BYTES (MB*DD*2)
#define BAND_LDS (SC_BYTES + QS_BYTES + 32*DD*2)

__global__ __launch_bounds__(384)
void k_band(const bf16_t* __restrict__ qb, const bf16_t* __restrict__ kb,
            const bf16_t* __restrict__ vb, float* __restrict__ ctx) {
    extern __shared__ char smem[];
    float (*sc)[3 * LW + 1] = (float(*)[3 * LW + 1])smem;
    bf16_t* Qs = (bf16_t*)(smem + SC_BYTES);
    bf16_t* Vt = (bf16_t*)(smem + SC_BYTES + QS_BYTES);

    const int blk = blockIdx.x;               // (LW/MB)*nch = 128
    const int h = blockIdx.y, b = blockIdx.z;
    const int c  = blk / (LW / MB);
    const int q0 = (blk % (LW / MB)) * MB;
    const int tid = threadIdx.x, wave = tid >> 5, lane = tid & 31;
    const int half = lane >> 4, lc = lane & 15;

    // stage Q rows (bf16): 384 x 16B chunks, one per thread
    {
        int r = tid / (DD / 8), cv = tid % (DD / 8);
        ((uint4*)Qs)[tid] = *(const uint4*)(qb + ((long)b * SPL + c * LW + q0 + r) * EE
                                            + h * DD + cv * 8);
    }
    __syncthreads();

    // ---- scores: wave w computes cols [w*64, w*64+64) ----
    {
        const int wn = wave * 64;
        v8f acc[4] = {};
#pragma unroll
        for (int ks = 0; ks < 6; ++ks) {
            v8bf lo = *(const v8bf*)&Qs[lc * DD + ks * 32 + 16 * half];
            v8bf hi = *(const v8bf*)&Qs[lc * DD + ks * 32 + 16 * half + 8];
            v16bf af = SHUF16(lo, hi);
#pragma unroll
            for (int nt = 0; nt < 4; ++nt) {
                int j = wn + nt * 16 + lc;
                int tk = c * LW + j - LW;
                tk = tk < 0 ? 0 : (tk >= SPL ? SPL - 1 : tk);   // clamp; masked below
                v16bf bfr = *(const v16bf*)(kb + ((long)b * SPL + tk) * EE
                                            + h * DD + ks * 32 + 16 * half);
                acc[nt] = __builtin_amdgcn_wmma_f32_16x16x32_bf16(
                    false, af, false, bfr, (short)0, acc[nt], false, false);
            }
        }
#pragma unroll
        for (int nt = 0; nt < 4; ++nt)
#pragma unroll
            for (int r = 0; r < 8; ++r)
                sc[half * 8 + r][1 + wn + nt * 16 + lc] = acc[nt][r];
    }
    __syncthreads();

    // band mask
    for (int i = tid; i < MB * (3 * LW); i += 384) {
        int r = i / (3 * LW), jj = i % (3 * LW);
        int qiw = q0 + r;
        int tk = c * LW + jj - LW;
        bool ok = (jj >= qiw) && (jj <= qiw + 2 * LW) && (tk >= 1) && (tk < SPL);
        if (!ok) sc[r][jj + 1] = -1e9f;
    }
    // gcol: q . k[token 0]
    for (int r = tid; r < MB; r += 384) {
        const bf16_t* qrow = Qs + r * DD;
        const bf16_t* k0 = kb + ((long)b * SPL) * EE + h * DD;
        float s = 0.f;
        for (int d = 0; d < DD; ++d) s = fmaf((float)qrow[d], (float)k0[d], s);
        sc[r][0] = s;
    }
    __syncthreads();

    // per-row softmax
    for (int r = wave; r < MB; r += 12) {
        float m = -1e30f;
        for (int j = lane; j < 3 * LW + 1; j += 32) m = fmaxf(m, sc[r][j]);
        for (int o = 16; o; o >>= 1) m = fmaxf(m, __shfl_xor(m, o, 32));
        float s = 0.f;
        for (int j = lane; j < 3 * LW + 1; j += 32) {
            float e2 = __expf(sc[r][j] - m); sc[r][j] = e2; s += e2;
        }
        for (int o = 16; o; o >>= 1) s += __shfl_xor(s, o, 32);
        float inv = 1.f / s;
        for (int j = lane; j < 3 * LW + 1; j += 32) sc[r][j] *= inv;
    }

    // ---- P*V via WMMA: wave w owns D cols [w*16, w*16+16) ----
    v8f accO = {};
    for (int ks2 = 0; ks2 < 24; ++ks2) {
        __syncthreads();
        // stage V tile (32 keys x 192 d) transposed: Vt[d*32 + kk]
        for (int j = tid; j < 32 * (DD / 8); j += 384) {
            int kk = j / (DD / 8), cv = j % (DD / 8);
            int tk = c * LW + ks2 * 32 + kk - LW;
            tk = tk < 0 ? 0 : (tk >= SPL ? SPL - 1 : tk);       // probs are 0 where masked
            v8bf tmp = *(const v8bf*)(vb + ((long)b * SPL + tk) * EE + h * DD + cv * 8);
#pragma unroll
            for (int t = 0; t < 8; ++t) Vt[(cv * 8 + t) * 32 + kk] = tmp[t];
        }
        __syncthreads();
        // A-frag: probabilities (f32->bf16) for keys [ks2*32, ks2*32+32)
        v16bf af;
#pragma unroll
        for (int e = 0; e < 16; ++e) {
            int kmap = e + 16 * half;
            af[e] = (bf16_t)sc[lc][1 + ks2 * 32 + kmap];
        }
        v16bf bfr = *(const v16bf*)(Vt + (wave * 16 + lc) * 32 + 16 * half);
        accO = __builtin_amdgcn_wmma_f32_16x16x32_bf16(
            false, af, false, bfr, (short)0, accO, false, false);
    }

    // epilogue: out = P_band*Vwin + p0 * v[token0]
    {
        int d = wave * 16 + lc;
        float v0 = (float)vb[((long)b * SPL) * EE + h * DD + d];
#pragma unroll
        for (int r = 0; r < 8; ++r) {
            int qi = half * 8 + r;
            float o = accO[r] + sc[qi][0] * v0;
            ctx[((long)b * SPL + c * LW + q0 + qi) * EE + h * DD + d] = o;
        }
    }
}

// ---------------------------------------------------------------------------
// Global-token attention (row 0), one block per (h,b)
// ---------------------------------------------------------------------------
__global__ __launch_bounds__(256)
void k_gattn(const float* __restrict__ qg, const float* __restrict__ kg,
             const float* __restrict__ vg, float* __restrict__ ctx) {
    const int h = blockIdx.x, b = blockIdx.y;
    __shared__ float sc[SPL];
    __shared__ float red[256];
    const float* qp = qg + (long)b * EE + h * DD;
    for (int t = threadIdx.x; t < SPL; t += 256)
        sc[t] = dot192f(qp, kg + ((long)b * SPL + t) * EE + h * DD);
    __syncthreads();
    float m = -1e30f;
    for (int t = threadIdx.x; t < SPL; t += 256) m = fmaxf(m, sc[t]);
    red[threadIdx.x] = m; __syncthreads();
    for (int o = 128; o; o >>= 1) {
        if (threadIdx.x < (unsigned)o) red[threadIdx.x] = fmaxf(red[threadIdx.x], red[threadIdx.x + o]);
        __syncthreads();
    }
    m = red[0]; __syncthreads();
    float s = 0.f;
    for (int t = threadIdx.x; t < SPL; t += 256) { float e2 = __expf(sc[t] - m); sc[t] = e2; s += e2; }
    red[threadIdx.x] = s; __syncthreads();
    for (int o = 128; o; o >>= 1) {
        if (threadIdx.x < (unsigned)o) red[threadIdx.x] += red[threadIdx.x + o];
        __syncthreads();
    }
    float inv = 1.f / red[0];
    __syncthreads();
    for (int d = threadIdx.x; d < DD; d += 256) {
        float a = 0.f;
        for (int t = 0; t < SPL; ++t)
            a = fmaf(sc[t], vg[((long)b * SPL + t) * EE + h * DD + d], a);
        ctx[((long)b * SPL) * EE + h * DD + d] = a * inv;
    }
}

// ===========================================================================
extern "C" void kernel_launch(void* const* d_in, const int* in_sizes, int n_in,
                              void* d_out, int out_size, void* d_ws, size_t ws_size,
                              hipStream_t stream) {
    (void)in_sizes; (void)n_in; (void)out_size; (void)ws_size;

    const float* feat    = (const float*)d_in[0];
    const float* cls     = (const float*)d_in[1];
    const float* pos_emb = (const float*)d_in[2];
    const float* tok_emb = (const float*)d_in[3];
    const float* emb_g   = (const float*)d_in[4];
    const float* emb_b   = (const float*)d_in[5];
    const float* hg      = (const float*)d_in[72];
    const float* hbv     = (const float*)d_in[73];
    const float* hw1     = (const float*)d_in[74];
    const float* hb1     = (const float*)d_in[75];
    const float* hw2     = (const float*)d_in[76];
    const float* hb2     = (const float*)d_in[77];
    const int*   pids    = (const int*)d_in[78];
    float* out = (float*)d_out;

    const size_t szF  = (size_t)NTOK * EE * 4;
    const size_t szB2 = (size_t)NTOK * EE * 2;
    char* ws = (char*)d_ws;
    float*  h    = (float*)(ws);
    bf16_t* hbf  = (bf16_t*)(ws + szF);
    bf16_t* qbB  = (bf16_t*)(ws + szF + szB2);
    bf16_t* kbB  = (bf16_t*)((char*)qbB + szB2);
    bf16_t* vbB  = (bf16_t*)((char*)kbB + szB2);
    float*  kgb  = (float*)((char*)vbB + szB2);
    float*  vgb  = (float*)((char*)kgb + szF);
    float*  ctx  = (float*)((char*)vgb + szF);
    bf16_t* ctxb = (bf16_t*)((char*)ctx + szF);
    bf16_t* wbuf = (bf16_t*)((char*)ctxb + szB2);
    float*  qgb  = (float*)((char*)wbuf + (size_t)EE * FFD * 2);
    float*  hA   = qgb + (size_t)BB * EE;
    float*  hB   = hA + (size_t)BB * EE;
    float*  proj = kgb;                  // alias: kg free after attention
    bf16_t* ff1b = (bf16_t*)vgb;         // alias: vg free after attention

    auto gemm = [&](const bf16_t* Ap, long lda, const float* Wsrc, const float* biasp,
                    float* Cp, bf16_t* Cbfp, int Mr, int Nc, int Kd, float alpha, int act) {
        long wn = (long)Kd * Nc;
        k_w_to_frag<<<dim3((int)((wn + 2047) / 2048)), dim3(256), 0, stream>>>(
            Wsrc, wbuf, Kd, Nc);
        k_wmma_gemm<<<dim3(Nc / 128, Mr / 128), dim3(256), 0, stream>>>(
            Ap, lda, wbuf, Nc / 128, biasp, Cp, Cbfp, Nc, Kd, alpha, act);
    };

    // Embedding + first LN
    {
        long n = (long)BB * SPL * EE;
        k_embed<<<dim3((int)((n + 255) / 256)), dim3(256), 0, stream>>>(
            feat, cls, pos_emb, tok_emb, pids, h);
        k_ln<<<dim3(NTOK), dim3(256), 0, stream>>>(
            h, EE, nullptr, 0, emb_g, emb_b, h, EE, hbf, EE, EE);
    }

    for (int li = 0; li < 3; ++li) {
        void* const* Lp = d_in + 6 + 22 * li;
        const float* wq  = (const float*)Lp[0];  const float* bq  = (const float*)Lp[1];
        const float* wk  = (const float*)Lp[2];  const float* bk  = (const float*)Lp[3];
        const float* wv  = (const float*)Lp[4];  const float* bv  = (const float*)Lp[5];
        const float* wo  = (const float*)Lp[6];  const float* bo  = (const float*)Lp[7];
        const float* wqg = (const float*)Lp[8];  const float* bqg = (const float*)Lp[9];
        const float* wkg = (const float*)Lp[10]; const float* bkg = (const float*)Lp[11];
        const float* wvg = (const float*)Lp[12]; const float* bvg = (const float*)Lp[13];
        const float* wi  = (const float*)Lp[14]; const float* bi  = (const float*)Lp[15];
        const float* wo2 = (const float*)Lp[16]; const float* bo2 = (const float*)Lp[17];
        const float* l1g = (const float*)Lp[18]; const float* l1b = (const float*)Lp[19];
        const float* l2g = (const float*)Lp[20]; const float* l2b = (const float*)Lp[21];

        gemm(hbf, EE, wq,  bq,  nullptr, qbB, NTOK, EE, EE, SCALEQ, 0);
        gemm(hbf, EE, wk,  bk,  nullptr, kbB, NTOK, EE, EE, 1.f,    0);
        gemm(hbf, EE, wv,  bv,  nullptr, vbB, NTOK, EE, EE, 1.f,    0);
        gemm(hbf, EE, wkg, bkg, kgb, nullptr, NTOK, EE, EE, 1.f,    0);
        gemm(hbf, EE, wvg, bvg, vgb, nullptr, NTOK, EE, EE, 1.f,    0);

        k_small_gemm<<<dim3((BB * EE + 255) / 256), dim3(256), 0, stream>>>(
            h, (long)SPL * EE, wqg, bqg, qgb, EE, BB, EE, EE, SCALEQ, 0);

        k_band<<<dim3((LW / MB) * (SPL / LW), HH, BB), dim3(384), BAND_LDS, stream>>>(
            qbB, kbB, vbB, ctx);
        k_gattn<<<dim3(HH, BB), dim3(256), 0, stream>>>(qgb, kgb, vgb, ctx);

        {
            long n = (long)NTOK * EE;
            k_f32_to_bf16<<<dim3((int)((n + 2047) / 2048)), dim3(256), 0, stream>>>(ctx, ctxb, n);
        }
        gemm(ctxb, EE, wo, bo, proj, nullptr, NTOK, EE, EE, 1.f, 0);
        k_ln<<<dim3(NTOK), dim3(256), 0, stream>>>(
            h, EE, proj, EE, l1g, l1b, h, EE, hbf, EE, EE);

        gemm(hbf, EE, wi, bi, nullptr, ff1b, NTOK, FFD, EE, 1.f, 1);
        gemm(ff1b, FFD, wo2, bo2, proj, nullptr, NTOK, EE, FFD, 1.f, 0);
        k_ln<<<dim3(NTOK), dim3(256), 0, stream>>>(
            h, EE, proj, EE, l2g, l2b, h, EE, hbf, EE, EE);
    }

    // Head
    k_ln<<<dim3(BB), dim3(256), 0, stream>>>(
        h, (long)SPL * EE, nullptr, 0, hg, hbv, hA, EE, nullptr, 0, EE);
    k_small_gemm<<<dim3((BB * EE + 255) / 256), dim3(256), 0, stream>>>(
        hA, EE, hw1, hb1, hB, EE, BB, EE, EE, 1.f, 1);
    k_small_gemm<<<dim3(1), dim3(256), 0, stream>>>(
        hB, EE, hw2, hb2, out, 1, BB, 1, EE, 1.f, 0);
}
